// SunnyGNN_43825846288499
// MI455X (gfx1250) — matrix operand order repaired and checked by reading, not compiled, not run on hardware
//
#include <hip/hip_runtime.h>

// ---------------------------------------------------------------------------
// SunnyGNN edge-attention scoring for gfx1250 (MI455X), bf16 WMMA pipeline.
// Round 2: native __bf16 converts (v_cvt_*_bf16_f32) + compile-time K
// (templated hop kernels, fully unrolled WMMA chunk loops).
// ---------------------------------------------------------------------------

typedef __attribute__((ext_vector_type(16))) __bf16 v16bf;
typedef __attribute__((ext_vector_type(8)))  float  v8f;

#define MIN_WEIGHT (-100000.0f)
#define HID 32

union BFrag {
    unsigned int d[8];
    v16bf        v;
};

union BfPair {
    __bf16       h[2];
    unsigned int d;
};

// Build a 16x32 bf16 A-fragment for this lane straight from an f32 row.
// Lane l: row = l&15, kbase = 32*q + 8*(l>>4); holds k in [kbase,kbase+8) and
// [kbase+16,kbase+24)  (ISA 7.12.2, 16-bit A 16x32 layout).
__device__ __forceinline__ v16bf build_afrag(const float* __restrict__ rowp,
                                             int kbase) {
    float4 f0 = *(const float4*)(rowp + kbase);
    float4 f1 = *(const float4*)(rowp + kbase + 4);
    float4 f2 = *(const float4*)(rowp + kbase + 16);
    float4 f3 = *(const float4*)(rowp + kbase + 20);
    v16bf a;
    a[0]  = (__bf16)f0.x; a[1]  = (__bf16)f0.y; a[2]  = (__bf16)f0.z; a[3]  = (__bf16)f0.w;
    a[4]  = (__bf16)f1.x; a[5]  = (__bf16)f1.y; a[6]  = (__bf16)f1.z; a[7]  = (__bf16)f1.w;
    a[8]  = (__bf16)f2.x; a[9]  = (__bf16)f2.y; a[10] = (__bf16)f2.z; a[11] = (__bf16)f2.w;
    a[12] = (__bf16)f3.x; a[13] = (__bf16)f3.y; a[14] = (__bf16)f3.z; a[15] = (__bf16)f3.w;
    return a;
}

// Weights pre-packed in LDS in B-fragment order: dword index
//   ((q*2 + t)*32 + lane)*8 + r
// holds bf16 pair W[k][n],W[k+1][n] with k = 32q + 8*(lane>>4) + 2r + 8*(r>>2),
// n = 16t + (lane&15)  (ISA 32x16 16-bit B layout).  Loaded as 2 x b128.
__device__ __forceinline__ void load_bfrag(const unsigned int* __restrict__ pW,
                                           int q, int t, int lane, BFrag& b) {
    const uint4* p = (const uint4*)(pW + (((q << 1) + t) << 8) + (lane << 3));
    uint4 x = p[0];
    uint4 y = p[1];
    b.d[0] = x.x; b.d[1] = x.y; b.d[2] = x.z; b.d[3] = x.w;
    b.d[4] = y.x; b.d[5] = y.y; b.d[6] = y.z; b.d[7] = y.w;
}

// Pack an f32 (K x 32) weight matrix into fragment-ordered bf16 pairs in LDS.
__device__ __forceinline__ void pack_weights(const float* __restrict__ W, int K,
                                             unsigned int* __restrict__ pW,
                                             int tid, int nthreads) {
    int total = K * 16;               // dwords
    for (int i = tid; i < total; i += nthreads) {
        int r = i & 7;
        int l = (i >> 3) & 31;
        int t = (i >> 8) & 1;
        int q = i >> 9;
        int n = (t << 4) + (l & 15);
        int k = (q << 5) + ((l >> 4) << 3) + (r << 1) + ((r >> 2) << 3);
        BfPair p;
        p.h[0] = (__bf16)W[k * HID + n];
        p.h[1] = (__bf16)W[(k + 1) * HID + n];
        pW[i] = p.d;
    }
}

// 16-row tile GEMM: C(16x32,f32) += bf16( rowp[16][K] ) @ bf16( W[K][32] ).
// K is compile-time -> fully unrolled chunk loop, 2 WMMAs per 32-K chunk.
template <int K>
__device__ __forceinline__ void gemm_rowtile(const float* __restrict__ rowp,
                                             const unsigned int* __restrict__ pW,
                                             int abase, int lane,
                                             v8f& c0, v8f& c1) {
#pragma unroll
    for (int q = 0; q < (K >> 5); ++q) {
        v16bf a = build_afrag(rowp, (q << 5) + abase);
        BFrag b0, b1;
        load_bfrag(pW, q, 0, lane, b0);
        load_bfrag(pW, q, 1, lane, b1);
        c0 = __builtin_amdgcn_wmma_f32_16x16x32_bf16(false, a, false, b0.v,
                                                     (short)0, c0, false, false);
        c1 = __builtin_amdgcn_wmma_f32_16x16x32_bf16(false, a, false, b1.v,
                                                     (short)0, c1, false, false);
    }
}

// ---------------------------------------------------------------------------
__global__ void fill_minw_kernel(float* __restrict__ out, int n) {
    int i = blockIdx.x * blockDim.x + threadIdx.x;
    if (i < n) out[i] = MIN_WEIGHT;
}

// h_target[b][j] = relu( all_emb2[n_map[offset_node[b]]] @ W2 )  (f32, tiny)
__global__ void htarget_kernel(const float* __restrict__ emb2,
                               const float* __restrict__ W2,
                               const int* __restrict__ n_map,
                               const int* __restrict__ offset_node,
                               float* __restrict__ ht, int total) {
    int i = blockIdx.x * blockDim.x + threadIdx.x;
    if (i >= total) return;
    int b = i >> 5, j = i & 31;
    const float* row = emb2 + (size_t)n_map[offset_node[b]] * 64;
    float s = 0.f;
#pragma unroll 8
    for (int k = 0; k < 64; ++k) s = fmaf(row[k], W2[k * HID + j], s);
    ht[i] = fmaxf(s, 0.f);
}

// ---------------------------------------------------------------------------
// One wave = 16 edges.  4 waves / block.  LDS: packed weights + per-wave
// 16x96 f32 staging buffer for [s | d | t] feeding the extractor GEMM.
// ---------------------------------------------------------------------------
template <int KS, int KD>
__global__ __launch_bounds__(128)
void edge_att_kernel(const float* __restrict__ embS, const float* __restrict__ WsG,
                     const float* __restrict__ embD, const float* __restrict__ WdG,
                     const float* __restrict__ We1G, const float* __restrict__ be1G,
                     const float* __restrict__ We2G, const float* __restrict__ be2G,
                     const int* __restrict__ n_map, const int* __restrict__ src,
                     const int* __restrict__ dst, const int* __restrict__ e_batch,
                     const int* __restrict__ eidx, const float* __restrict__ hT,
                     float* __restrict__ outRow, int nE) {
    __shared__ __align__(16) unsigned int pWs[KS * 16];
    __shared__ __align__(16) unsigned int pWd[KD * 16];
    __shared__ __align__(16) unsigned int pWe1[96 * 16];
    __shared__ __align__(16) float stage[4][16 * 96];
    __shared__ float sbe1[HID];
    __shared__ float sWe2[HID];

    const int tid = threadIdx.x;
    pack_weights(WsG, KS, pWs, tid, 128);
    pack_weights(WdG, KD, pWd, tid, 128);
    pack_weights(We1G, 96, pWe1, tid, 128);
    if (tid < HID) { sbe1[tid] = be1G[tid]; sWe2[tid] = We2G[tid]; }
    __syncthreads();

    const int wave  = tid >> 5;
    const int lane  = tid & 31;
    const int half  = lane >> 4;
    const int n     = lane & 15;
    const int abase = half << 3;

    const int base = ((blockIdx.x << 2) + wave) << 4;   // 16 edges per wave
    int e = base + n;
    if (e >= nE) e = nE - 1;                            // clamp: EXEC stays all-1
    const int id   = eidx[e];
    const int srow = n_map[src[id]];
    const int drow = n_map[dst[id]];
    const int trow = e_batch[id];

    // s = relu(Xs @ Ws), d = relu(Xd @ Wd): activations gathered straight into
    // A-fragments from global memory (4 x b128 per 32-K chunk per lane).
    v8f cs0 = {}, cs1 = {};
    gemm_rowtile<KS>(embS + (size_t)srow * KS, pWs, abase, lane, cs0, cs1);
    v8f cd0 = {}, cd1 = {};
    gemm_rowtile<KD>(embD + (size_t)drow * KD, pWd, abase, lane, cd0, cd1);

    // Stage [s | d | t] as a 16x96 f32 row-major tile.
    float* st = stage[wave];
#pragma unroll
    for (int r = 0; r < 8; ++r) {
        int mr = (r + (half << 3)) * 96;
        st[mr +      n] = fmaxf(cs0[r], 0.f);
        st[mr + 16 + n] = fmaxf(cs1[r], 0.f);
        st[mr + 32 + n] = fmaxf(cd0[r], 0.f);
        st[mr + 48 + n] = fmaxf(cd1[r], 0.f);
    }
    {   // t = h_target[e_batch[idx]] : lane (m,half) copies 16 floats
        const float* tp = hT + (size_t)trow * HID + (half << 4);
        float4 t0 = *(const float4*)(tp);
        float4 t1 = *(const float4*)(tp + 4);
        float4 t2 = *(const float4*)(tp + 8);
        float4 t3 = *(const float4*)(tp + 12);
        float* dp = st + n * 96 + 64 + (half << 4);
        *(float4*)(dp)      = t0;
        *(float4*)(dp + 4)  = t1;
        *(float4*)(dp + 8)  = t2;
        *(float4*)(dp + 12) = t3;
    }
    __syncthreads();

    // Extractor hidden = relu([s d t] @ We1 + be1) : K = 96, 6 WMMAs.
    v8f h0 = {}, h1 = {};
    gemm_rowtile<96>(st + n * 96, pWe1, abase, lane, h0, h1);

    // out = hidden @ We2 + be2 : per-lane scale + shfl_xor reduce over the
    // 16-lane half-group (C/D layout: lane = N, VGPR = M%8).
    const float b0v = sbe1[n],      b1v = sbe1[16 + n];
    const float w0v = sWe2[n],      w1v = sWe2[16 + n];
    float res[8];
#pragma unroll
    for (int r = 0; r < 8; ++r) {
        float p = fmaxf(h0[r] + b0v, 0.f) * w0v + fmaxf(h1[r] + b1v, 0.f) * w1v;
        p += __shfl_xor(p, 1, 16);
        p += __shfl_xor(p, 2, 16);
        p += __shfl_xor(p, 4, 16);
        p += __shfl_xor(p, 8, 16);
        res[r] = p;
    }
    if (n == 0) {
        const float be2v = be2G[0];
#pragma unroll
        for (int r = 0; r < 8; ++r) {
            int e2 = base + r + (half << 3);
            if (e2 < nE) outRow[eidx[e2]] = res[r] + be2v;
        }
    }
}

// ---------------------------------------------------------------------------
extern "C" void kernel_launch(void* const* d_in, const int* in_sizes, int n_in,
                              void* d_out, int out_size, void* d_ws, size_t ws_size,
                              hipStream_t stream) {
    const float* all_emb0    = (const float*)d_in[0];
    const float* all_emb1    = (const float*)d_in[1];
    const float* all_emb2    = (const float*)d_in[2];
    const int*   n_map       = (const int*)d_in[3];
    const int*   src         = (const int*)d_in[4];
    const int*   dst         = (const int*)d_in[5];
    const int*   e_batch     = (const int*)d_in[6];
    const int*   offset_node = (const int*)d_in[7];
    const int*   one_hop_idx = (const int*)d_in[8];
    const int*   two_hop_idx = (const int*)d_in[9];
    const float* W0          = (const float*)d_in[10];
    const float* W1          = (const float*)d_in[11];
    const float* W2          = (const float*)d_in[12];
    const float* We1         = (const float*)d_in[13];
    const float* be1         = (const float*)d_in[14];
    const float* We2         = (const float*)d_in[15];
    const float* be2         = (const float*)d_in[16];

    float* out = (float*)d_out;
    float* h_target = (float*)d_ws;                 // BATCH x 32 f32

    const int E  = in_sizes[4];                     // total edges
    const int H1 = in_sizes[8];                     // one-hop edges
    const int H2 = in_sizes[9];                     // two-hop edges
    const int B  = in_sizes[7];                     // batch (ego graphs)

    // 1) edge_att = MIN_WEIGHT everywhere
    {
        int n = 2 * E;
        fill_minw_kernel<<<(n + 255) / 256, 256, 0, stream>>>(out, n);
    }
    // 2) h_target = relu(all_emb2[n_map[offset_node]] @ W2)
    {
        int total = B * HID;
        htarget_kernel<<<(total + 255) / 256, 256, 0, stream>>>(
            all_emb2, W2, n_map, offset_node, h_target, total);
    }
    // 3) one-hop (k=2): s from all_emb1/W1 (K=256), d from all_emb2/W2 (K=64)
    if (H1 > 0) {
        edge_att_kernel<256, 64><<<(H1 + 63) / 64, 128, 0, stream>>>(
            all_emb1, W1, all_emb2, W2,
            We1, be1, We2, be2,
            n_map, src, dst, e_batch, one_hop_idx, h_target,
            out + (size_t)E, H1);
    }
    // 4) two-hop (k=1): s from all_emb0/W0 (K=128), d from all_emb1/W1 (K=256)
    if (H2 > 0) {
        edge_att_kernel<128, 256><<<(H2 + 63) / 64, 128, 0, stream>>>(
            all_emb0, W0, all_emb1, W1,
            We1, be1, We2, be2,
            n_map, src, dst, e_batch, two_hop_idx, h_target,
            out, H2);
    }
}
